// FCOS_13898514170668
// MI455X (gfx1250) — matrix operand order repaired and checked by reading, not compiled, not run on hardware
//
#include <hip/hip_runtime.h>
#include <stdint.h>

typedef __bf16 bf16_t;
typedef __attribute__((ext_vector_type(16))) __bf16 v16bf;
typedef __attribute__((ext_vector_type(8)))  float  v8f;

#define CDIM 256
#define KB_CNT 8   // 256 / 32

__device__ __forceinline__ unsigned short f2bf(float f) {
  unsigned int u = __float_as_uint(f);
  u += 0x7FFFu + ((u >> 16) & 1u);          // round-to-nearest-even
  return (unsigned short)(u >> 16);
}

union AFrag { uint4 q[2]; v16bf v; };

// ---------------------------------------------------------------------------
// Zero a buffer with 16B stores (halo init for padded bf16 buffers)
// ---------------------------------------------------------------------------
__global__ void __launch_bounds__(256)
k_zero_u4(uint4* __restrict__ p, size_t n16) {
  size_t i = (size_t)blockIdx.x * 256 + threadIdx.x;
  if (i < n16) p[i] = make_uint4(0, 0, 0, 0);
}

__global__ void __launch_bounds__(256)
k_zero(float* p, int n) { int i = blockIdx.x * 256 + threadIdx.x; if (i < n) p[i] = 0.f; }

// ---------------------------------------------------------------------------
// fp32 NCHW -> bf16 NHWC with 1-pixel zero halo.
// block (32,8), grid (HW/32, C/32, N). Interior writes only.
// ---------------------------------------------------------------------------
__global__ void __launch_bounds__(256)
k_nchw_to_nhwc_pad(const float* __restrict__ in, bf16_t* __restrict__ out,
                   int HW, int W, int WP, int PHW) {
  __shared__ float tile[32][33];
  const int n  = blockIdx.z;
  const int p0 = blockIdx.x * 32;
  const int c0 = blockIdx.y * 32;
  const int tx = threadIdx.x, ty = threadIdx.y;
#pragma unroll
  for (int i = 0; i < 32; i += 8)
    tile[ty + i][tx] = in[((size_t)n * CDIM + c0 + ty + i) * HW + p0 + tx];
  __syncthreads();
#pragma unroll
  for (int i = 0; i < 32; i += 8) {
    const int p = p0 + ty + i;
    const int y = p / W, x = p % W;
    ((unsigned short*)out)[((size_t)n * PHW + (size_t)(y + 1) * WP + (x + 1)) * CDIM
                           + c0 + tx] = f2bf(tile[tx][ty + i]);
  }
}

// ---------------------------------------------------------------------------
// Pack OIHW fp32 weights -> per-lane B-fragment order (bf16 pairs).
// dword index = (((t*NB + nb)*8 + kb)*32 + lane)*8 + v
// ---------------------------------------------------------------------------
__global__ void __launch_bounds__(256)
k_pack_w(const float* __restrict__ w1, int c1, const float* __restrict__ w2, int c2,
         uint32_t* __restrict__ dst, int NB) {
  int idx = blockIdx.x * 256 + threadIdx.x;
  int total = 9 * NB * KB_CNT * 32 * 8;
  if (idx >= total) return;
  int v = idx & 7;
  int r = idx >> 3;
  int lane = r & 31; r >>= 5;
  int kb = r & 7;    r >>= 3;
  int nb = r % NB;
  int t  = r / NB;
  int N   = nb * 16 + (lane & 15);
  int cin = kb * 32 + ((v >> 2) ? 16 : 0) + ((lane >> 4) ? 8 : 0) + (v & 3) * 2;
  int ky = t / 3, kx = t % 3;
  float f0 = 0.f, f1 = 0.f;
  if (N < c1) {
    const float* w = w1 + (size_t)N * CDIM * 9;
    f0 = w[(size_t)cin * 9 + ky * 3 + kx];
    f1 = w[(size_t)(cin + 1) * 9 + ky * 3 + kx];
  } else if (N < c1 + c2) {
    const float* w = w2 + (size_t)(N - c1) * CDIM * 9;
    f0 = w[(size_t)cin * 9 + ky * 3 + kx];
    f1 = w[(size_t)(cin + 1) * 9 + ky * 3 + kx];
  }
  dst[idx] = (uint32_t)f2bf(f0) | ((uint32_t)f2bf(f1) << 16);
}

// ---------------------------------------------------------------------------
// Implicit-GEMM 3x3 conv on zero-padded bf16 NHWC input; bf16 WMMA, f32 acc.
// One wave: 16(pos) x (NT*16)(cout) tile; 8 waves/block over position tiles.
// Two-stage software pipeline over the 72-step (tap x kb) K loop: loads for
// stage s+1 are in flight while WMMAs consume stage s (partial loadcnt waits).
// mode 0: fp32 NHWC scratch (+bias); mode 1: cls logits; mode 2: reg(relu)+ctr
// ---------------------------------------------------------------------------
template <int NT>
__global__ void __launch_bounds__(256)
k_conv3x3_wmma(const bf16_t* __restrict__ xin, const uint32_t* __restrict__ wpack,
               int W, int WP, int HW, int PHW, int NB, int Cvalid,
               const float* __restrict__ bias, const float* __restrict__ bias2,
               int mode, float* __restrict__ out, float* __restrict__ ctrout,
               int lvloff, int Ptot) {
  const int lane = threadIdx.x & 31;
  const int wave = threadIdx.x >> 5;
  const int half = lane >> 4;
  const int PT   = HW >> 4;
  const int ptile = blockIdx.x * 8 + wave;
  if (ptile >= PT) return;                    // wave-uniform exit
  const int nb0 = blockIdx.y * NT;
  const int n   = blockIdx.z;
  const int p   = ptile * 16 + (lane & 15);   // this lane's M position
  const int y = p / W, x = p % W;
  // tap (dy,dx) in 0..2 reads padded coord (y+dy, x+dx); origin shift by +1
  const bf16_t* aptr =
      xin + ((size_t)n * PHW + (size_t)y * WP + x) * CDIM + half * 8;
  const uint32_t* bbase = wpack + (size_t)nb0 * (KB_CNT * 256) + lane * 8;

  auto aload = [&](int s, AFrag& a) {
    const int t = s >> 3, kb = s & 7;
    const int dy = t / 3, dx = t % 3;
    const uint4* pa = (const uint4*)(aptr + ((size_t)dy * WP + dx) * CDIM + kb * 32);
    a.q[0] = pa[0];                           // ch +0..7  (half=0) / +8..15 (half=1)
    a.q[1] = pa[2];                           // ch +16..23 / +24..31
  };
  auto bload = [&](int s, AFrag* b) {
    const int t = s >> 3, kb = s & 7;
    const uint32_t* brow = bbase + (size_t)t * NB * (KB_CNT * 256) + kb * 256;
#pragma unroll
    for (int j = 0; j < NT; ++j) {
      const uint4* pb = (const uint4*)(brow + (size_t)j * (KB_CNT * 256));
      b[j].q[0] = pb[0];
      b[j].q[1] = pb[1];
    }
  };

  v8f zero = {};
  v8f acc[NT];
#pragma unroll
  for (int j = 0; j < NT; ++j) acc[j] = zero;

  AFrag a[2];
  AFrag b[2][NT];
  aload(0, a[0]);
  bload(0, b[0]);

  const int STEPS = 9 * KB_CNT;               // 72
#pragma unroll 1
  for (int s = 0; s < STEPS; s += 2) {
    // stage s executes; prefetch stage s+1 into buffer 1
    aload(s + 1, a[1]);
    bload(s + 1, b[1]);
#pragma unroll
    for (int j = 0; j < NT; ++j)
      acc[j] = __builtin_amdgcn_wmma_f32_16x16x32_bf16(
          false, a[0].v, false, b[0][j].v, (short)0, acc[j], false, false);
    // stage s+1 executes; prefetch stage s+2 into buffer 0
    if (s + 2 < STEPS) {
      aload(s + 2, a[0]);
      bload(s + 2, b[0]);
    }
#pragma unroll
    for (int j = 0; j < NT; ++j)
      acc[j] = __builtin_amdgcn_wmma_f32_16x16x32_bf16(
          false, a[1].v, false, b[1][j].v, (short)0, acc[j], false, false);
  }

  const int ncol = lane & 15;
#pragma unroll
  for (int j = 0; j < NT; ++j) {
    const int cout = (nb0 + j) * 16 + ncol;
    if (cout >= Cvalid) continue;
    float bv;
    if (mode == 2) bv = (cout < 4) ? bias[cout] : (cout == 4 ? bias2[0] : 0.f);
    else           bv = bias ? bias[cout] : 0.f;
#pragma unroll
    for (int r = 0; r < 8; ++r) {             // D layout: VGPR r -> M = r + half*8
      const int pp = ptile * 16 + r + half * 8;
      float v = acc[j][r] + bv;
      if (mode == 0) {
        out[((size_t)n * HW + pp) * CDIM + cout] = v;
      } else if (mode == 1) {
        out[((size_t)n * Ptot + lvloff + pp) * 20 + cout] = v;
      } else {
        if (cout < 4)       out[((size_t)n * Ptot + lvloff + pp) * 4 + cout] = fmaxf(v, 0.f);
        else if (cout == 4) ctrout[(size_t)n * Ptot + lvloff + pp] = v;
      }
    }
  }
}

// ---------------------------------------------------------------------------
// GroupNorm support
// ---------------------------------------------------------------------------
__global__ void __launch_bounds__(256)
k_gn_stats(const float* __restrict__ xf, float* __restrict__ stats, int HW) {
  const int ng = blockIdx.x;                  // n*16 + g
  const int n = ng >> 4, g = ng & 15;
  const float* base = xf + (size_t)n * HW * CDIM + g * 16;
  float s = 0.f, s2 = 0.f;
  const int stride = gridDim.y * 256;
  for (int p = blockIdx.y * 256 + threadIdx.x; p < HW; p += stride) {
    const float* row = base + (size_t)p * CDIM;
#pragma unroll
    for (int c = 0; c < 16; ++c) { float v = row[c]; s += v; s2 += v * v; }
  }
  __shared__ float rs[256], rq[256];
  rs[threadIdx.x] = s; rq[threadIdx.x] = s2;
  __syncthreads();
  for (int o = 128; o > 0; o >>= 1) {
    if ((int)threadIdx.x < o) { rs[threadIdx.x] += rs[threadIdx.x + o];
                                rq[threadIdx.x] += rq[threadIdx.x + o]; }
    __syncthreads();
  }
  if (threadIdx.x == 0) {
    atomicAdd(&stats[ng * 2 + 0], rs[0]);
    atomicAdd(&stats[ng * 2 + 1], rq[0]);
  }
}

// normalize + affine + ReLU + bf16 repack into padded NHWC (interior only)
__global__ void __launch_bounds__(256)
k_gn_norm_relu(const float* __restrict__ xf, bf16_t* __restrict__ xo,
               const float* __restrict__ stats, const float* __restrict__ gamma,
               const float* __restrict__ beta, int HW, int W, int WP, int PHW) {
  size_t idx = (size_t)blockIdx.x * 256 + threadIdx.x;
  size_t total = (size_t)8 * HW * CDIM;
  if (idx >= total) return;
  const int c = (int)(idx & 255);
  const size_t np = idx >> 8;                 // n*HW + p
  const int n = (int)(np / (size_t)HW);
  const int p = (int)(np - (size_t)n * HW);
  const int g = c >> 4;
  const float cnt = 16.0f * (float)HW;
  const float m  = stats[(n * 16 + g) * 2 + 0] / cnt;
  const float vv = stats[(n * 16 + g) * 2 + 1] / cnt - m * m;
  const float r  = rsqrtf(vv + 1e-5f);
  float val = (xf[idx] - m) * r * gamma[c] + beta[c];
  val = fmaxf(val, 0.0f);
  const int y = p / W, x = p % W;
  ((unsigned short*)xo)[((size_t)n * PHW + (size_t)(y + 1) * WP + (x + 1)) * CDIM + c]
      = f2bf(val);
}

// ---------------------------------------------------------------------------
extern "C" void kernel_launch(void* const* d_in, const int* in_sizes, int n_in,
                              void* d_out, int out_size, void* d_ws, size_t ws_size,
                              hipStream_t stream) {
  (void)in_sizes; (void)n_in; (void)out_size; (void)ws_size;
  const float* feat[3]    = {(const float*)d_in[0], (const float*)d_in[1], (const float*)d_in[2]};
  const float* cls_conv_w = (const float*)d_in[3];
  const float* cls_conv_b = (const float*)d_in[4];
  const float* cls_gn_g   = (const float*)d_in[5];
  const float* cls_gn_b   = (const float*)d_in[6];
  const float* cls_out_w  = (const float*)d_in[7];
  const float* cls_out_b  = (const float*)d_in[8];
  const float* reg_conv_w = (const float*)d_in[9];
  const float* reg_conv_b = (const float*)d_in[10];
  const float* reg_gn_g   = (const float*)d_in[11];
  const float* reg_gn_b   = (const float*)d_in[12];
  const float* reg_out_w  = (const float*)d_in[13];
  const float* reg_out_b  = (const float*)d_in[14];
  const float* ctr_w      = (const float*)d_in[15];
  const float* ctr_b      = (const float*)d_in[16];

  char* ws = (char*)d_ws;
  size_t off = 0;
  auto alloc = [&](size_t bytes) {
    char* p = ws + off;
    off += (bytes + 255) & ~(size_t)255;
    return (void*)p;
  };
  const size_t packCC = (size_t)9 * 16 * KB_CNT * 32 * 8 * 4;
  uint32_t* wpCls0   = (uint32_t*)alloc(packCC);
  uint32_t* wpCls1   = (uint32_t*)alloc(packCC);
  uint32_t* wpReg0   = (uint32_t*)alloc(packCC);
  uint32_t* wpReg1   = (uint32_t*)alloc(packCC);
  uint32_t* wpClsOut = (uint32_t*)alloc((size_t)9 * 2 * KB_CNT * 32 * 8 * 4);
  uint32_t* wpRegCtr = (uint32_t*)alloc((size_t)9 * 1 * KB_CNT * 32 * 8 * 4);
  float*  stats = (float*)alloc(256 * 4);
  const size_t MAXPHW = 102 * 130;            // level-0 padded spatial size
  bf16_t* xin   = (bf16_t*)alloc((size_t)8 * MAXPHW * CDIM * 2);
  bf16_t* a1    = (bf16_t*)alloc((size_t)8 * MAXPHW * CDIM * 2);
  bf16_t* a2    = (bf16_t*)alloc((size_t)8 * MAXPHW * CDIM * 2);
  float*  tf32  = (float*)alloc((size_t)8 * 12800 * CDIM * 4);

  // ---- pack all weights once per launch ----
  {
    int tot = 9 * 16 * KB_CNT * 32 * 8, blk = (tot + 255) / 256;
    k_pack_w<<<blk, 256, 0, stream>>>(cls_conv_w,                 256, nullptr, 0, wpCls0, 16);
    k_pack_w<<<blk, 256, 0, stream>>>(cls_conv_w + 256 * 256 * 9, 256, nullptr, 0, wpCls1, 16);
    k_pack_w<<<blk, 256, 0, stream>>>(reg_conv_w,                 256, nullptr, 0, wpReg0, 16);
    k_pack_w<<<blk, 256, 0, stream>>>(reg_conv_w + 256 * 256 * 9, 256, nullptr, 0, wpReg1, 16);
    int t2 = 9 * 2 * KB_CNT * 32 * 8;
    k_pack_w<<<(t2 + 255) / 256, 256, 0, stream>>>(cls_out_w, 20, nullptr, 0, wpClsOut, 2);
    int t1 = 9 * 1 * KB_CNT * 32 * 8;
    k_pack_w<<<(t1 + 255) / 256, 256, 0, stream>>>(reg_out_w, 4, ctr_w, 1, wpRegCtr, 1);
  }

  const int Hs[3] = {100, 50, 25}, Wd[3] = {128, 64, 32};
  const int HWs[3] = {12800, 3200, 800};
  const int lvloff[3] = {0, 12800, 16000};
  const int Ptot = 16800;
  float* outCls = (float*)d_out;
  float* outReg = outCls + (size_t)8 * Ptot * 20;
  float* outCtr = outReg + (size_t)8 * Ptot * 4;

  for (int L = 0; L < 3; ++L) {
    const int H = Hs[L], W = Wd[L], HW = HWs[L];
    const int WP = W + 2, HP = H + 2, PHW = HP * WP;
    const int PT = HW / 16;

    // zero padded bf16 buffers (halo must be 0; interiors get overwritten)
    {
      size_t n16 = ((size_t)8 * PHW * CDIM * 2 + 15) / 16;
      int zb = (int)((n16 + 255) / 256);
      k_zero_u4<<<zb, 256, 0, stream>>>((uint4*)xin, n16);
      k_zero_u4<<<zb, 256, 0, stream>>>((uint4*)a1, n16);
      k_zero_u4<<<zb, 256, 0, stream>>>((uint4*)a2, n16);
    }

    dim3 tb(32, 8, 1), tg(HW / 32, CDIM / 32, 8);
    k_nchw_to_nhwc_pad<<<tg, tb, 0, stream>>>(feat[L], xin, HW, W, WP, PHW);

    dim3 cb(256, 1, 1);
    dim3 cg((PT + 7) / 8, 4, 8);              // NT=4 -> 4 cout blocks of 64
    dim3 sg(128, 8, 1);
    size_t tot = (size_t)8 * HW * CDIM;
    int nblk = (int)((tot + 255) / 256);

    // ---- classification tower ----
    k_conv3x3_wmma<4><<<cg, cb, 0, stream>>>(xin, wpCls0, W, WP, HW, PHW, 16, 256,
        cls_conv_b, nullptr, 0, tf32, nullptr, 0, Ptot);
    k_zero<<<1, 256, 0, stream>>>(stats, 256);
    k_gn_stats<<<sg, 256, 0, stream>>>(tf32, stats, HW);
    k_gn_norm_relu<<<nblk, 256, 0, stream>>>(tf32, a1, stats, cls_gn_g, cls_gn_b,
                                             HW, W, WP, PHW);

    k_conv3x3_wmma<4><<<cg, cb, 0, stream>>>(a1, wpCls1, W, WP, HW, PHW, 16, 256,
        cls_conv_b + 256, nullptr, 0, tf32, nullptr, 0, Ptot);
    k_zero<<<1, 256, 0, stream>>>(stats, 256);
    k_gn_stats<<<sg, 256, 0, stream>>>(tf32, stats, HW);
    k_gn_norm_relu<<<nblk, 256, 0, stream>>>(tf32, a2, stats, cls_gn_g + 256, cls_gn_b + 256,
                                             HW, W, WP, PHW);

    dim3 og((PT + 7) / 8, 1, 8);
    k_conv3x3_wmma<2><<<og, cb, 0, stream>>>(a2, wpClsOut, W, WP, HW, PHW, 2, 20,
        cls_out_b, nullptr, 1, outCls, nullptr, lvloff[L], Ptot);

    // ---- regression tower ----
    k_conv3x3_wmma<4><<<cg, cb, 0, stream>>>(xin, wpReg0, W, WP, HW, PHW, 16, 256,
        reg_conv_b, nullptr, 0, tf32, nullptr, 0, Ptot);
    k_zero<<<1, 256, 0, stream>>>(stats, 256);
    k_gn_stats<<<sg, 256, 0, stream>>>(tf32, stats, HW);
    k_gn_norm_relu<<<nblk, 256, 0, stream>>>(tf32, a1, stats, reg_gn_g, reg_gn_b,
                                             HW, W, WP, PHW);

    k_conv3x3_wmma<4><<<cg, cb, 0, stream>>>(a1, wpReg1, W, WP, HW, PHW, 16, 256,
        reg_conv_b + 256, nullptr, 0, tf32, nullptr, 0, Ptot);
    k_zero<<<1, 256, 0, stream>>>(stats, 256);
    k_gn_stats<<<sg, 256, 0, stream>>>(tf32, stats, HW);
    k_gn_norm_relu<<<nblk, 256, 0, stream>>>(tf32, a2, stats, reg_gn_g + 256, reg_gn_b + 256,
                                             HW, W, WP, PHW);

    k_conv3x3_wmma<1><<<og, cb, 0, stream>>>(a2, wpRegCtr, W, WP, HW, PHW, 1, 5,
        reg_out_b, ctr_b, 2, outReg, outCtr, lvloff[L], Ptot);
  }
}
